// TreeTextEmbed_31550829756813
// MI455X (gfx1250) — matrix operand order repaired and checked by reading, not compiled, not run on hardware
//
#include <hip/hip_runtime.h>

// ---------------------------------------------------------------------------
// TreeTextEmbed for MI455X (gfx1250): bf16 WMMA GEMMs + fused elementwise.
// B=32, S=4096, D=512, H=8, HD=64, L=512
// ---------------------------------------------------------------------------

typedef __bf16 v16bf __attribute__((ext_vector_type(16)));
typedef __bf16 v8bf  __attribute__((ext_vector_type(8)));
typedef float  v8f   __attribute__((ext_vector_type(8)));

#define BN  32
#define SN  4096
#define DN  512
#define HN  8
#define HDN 64
#define LN_ 512
#define NQ  1536

__device__ __forceinline__ unsigned short f2bf(float f) {
  unsigned u = __builtin_bit_cast(unsigned, f);
  u += 0x7fffu + ((u >> 16) & 1u);   // round-to-nearest-even
  return (unsigned short)(u >> 16);
}

// --------------------------- f32 -> bf16 convert ---------------------------
__global__ void k_cvt_bf16(const float* __restrict__ a,
                           unsigned short* __restrict__ o, int n) {
  for (int i = blockIdx.x * blockDim.x + threadIdx.x; i < n;
       i += gridDim.x * blockDim.x)
    o[i] = f2bf(a[i]);
}

// --------------- embed: ids*id_w + id_b + pe, LN, pool pairs ---------------
// grid = B * (S/2) blocks, 256 threads (2 dims/thread/position)
__global__ void k_embed_pool2(const int* __restrict__ text,
                              const float* __restrict__ pe,
                              const float* __restrict__ idw,
                              const float* __restrict__ idb,
                              const float* __restrict__ eg,
                              const float* __restrict__ eb,
                              float* __restrict__ outf,
                              unsigned short* __restrict__ outb) {
  __shared__ float rs[256], rq[256];
  int R = blockIdx.x;                 // pooled row in [0, B*2048)
  int b = R >> 11, pp = R & 2047;
  int t = threadIdx.x;
  float a0 = 0.f, a1 = 0.f;
  for (int j = 0; j < 2; ++j) {
    int s = pp * 2 + j;
    float tv = (float)text[b * SN + s];
    float e0 = tv * idw[t]       + idb[t]       + pe[s * DN + t];
    float e1 = tv * idw[t + 256] + idb[t + 256] + pe[s * DN + t + 256];
    rs[t] = e0 + e1; rq[t] = e0 * e0 + e1 * e1;
    __syncthreads();
    for (int st = 128; st > 0; st >>= 1) {
      if (t < st) { rs[t] += rs[t + st]; rq[t] += rq[t + st]; }
      __syncthreads();
    }
    float m = rs[0] * (1.f / DN);
    float var = rq[0] * (1.f / DN) - m * m;
    float inv = rsqrtf(var + 1e-5f);
    __syncthreads();
    a0 += 0.5f * ((e0 - m) * inv * eg[t]       + eb[t]);
    a1 += 0.5f * ((e1 - m) * inv * eg[t + 256] + eb[t + 256]);
  }
  size_t o = (size_t)R * DN;
  outf[o + t] = a0;            outf[o + t + 256] = a1;
  outb[o + t] = f2bf(a0);      outb[o + t + 256] = f2bf(a1);
}

// ------------------------------ WMMA GEMM ----------------------------------
// C[M,N] = A[M,K](bf16) * W[N,K](bf16)^T + bias[N]   (f32 accumulate)
// block = 256 thr = 8 waves; block tile 64(M) x 128(N); wave tile 64 x 16.
// Double-buffered LDS A tile (2 x 64x32 bf16): one barrier per K-step; the
// next tile's global load is issued before the WMMA block so it overlaps.
__global__ void k_gemm_bf16(const unsigned short* __restrict__ A,
                            const unsigned short* __restrict__ W,
                            const float* __restrict__ bias,
                            float* __restrict__ C, int M, int N, int K) {
  __shared__ unsigned short smA[2][64 * 32];
  int t = threadIdx.x;
  int lane = t & 31, wave = t >> 5;
  int mbase = blockIdx.y * 64;
  int nb = blockIdx.x * 128 + wave * 16;
  int r = lane & 15, hf = lane >> 4;

  v8f acc[4];
#pragma unroll
  for (int mt = 0; mt < 4; ++mt) acc[mt] = (v8f){0,0,0,0,0,0,0,0};

  int crow = t >> 2, cpart = t & 3;   // cooperative A-staging assignment
  const unsigned short* aptr = &A[(size_t)(mbase + crow) * K + cpart * 8];
  int sslot = crow * 32 + cpart * 8;

  // prologue: stage tile 0
  *(uint4*)&smA[0][sslot] = *(const uint4*)aptr;
  __syncthreads();

  for (int kb = 0; kb < K; kb += 32) {
    int cur = (kb >> 5) & 1;
    bool more = (kb + 32) < K;
    uint4 anext;
    if (more) {
      anext = *(const uint4*)(aptr + kb + 32);     // tile k+1 -> regs (in flight)
      __builtin_prefetch(aptr + kb + 64, 0, 3);    // tile k+2 -> near cache
    }
    // B operand: lane n = r holds W row (nb+r), contiguous K chunk of 16
    v16bf bfrag =
        *(const v16bf*)((const __bf16*)&W[(size_t)(nb + r) * K + kb + hf * 16]);

#pragma unroll
    for (int mt = 0; mt < 4; ++mt) {
      // A operand, contiguous K order matching B: lane half -> K 0-15/16-31
      const __bf16* ap = (const __bf16*)&smA[cur][(mt * 16 + r) * 32 + hf * 16];
      v8bf lo = *(const v8bf*)ap;
      v8bf hi = *(const v8bf*)(ap + 8);
      v16bf af;
#pragma unroll
      for (int i = 0; i < 8; ++i) { af[i] = lo[i]; af[i + 8] = hi[i]; }
      acc[mt] = __builtin_amdgcn_wmma_f32_16x16x32_bf16(
          false, af, false, bfrag, (short)0, acc[mt], false, false);
    }

    if (more) *(uint4*)&smA[cur ^ 1][sslot] = anext;
    __syncthreads();   // writes of buf[cur^1] visible; reads of buf[cur] done
  }

  int col = nb + r;
  float bs = bias[col];
#pragma unroll
  for (int mt = 0; mt < 4; ++mt)
#pragma unroll
    for (int i = 0; i < 8; ++i)
      C[(size_t)(mbase + mt * 16 + hf * 8 + i) * N + col] = acc[mt][i] + bs;
}

// ------------- LIF gate + residual + LN + optional pool-by-G ---------------
// grid = B * (Lin/G) blocks; y = proj*(proj>=th) + fused ; LN ; mean over G
__global__ void k_lif_ln_pool(const float* __restrict__ proj,
                              const float* __restrict__ fused,
                              const float* __restrict__ th,
                              const float* __restrict__ g,
                              const float* __restrict__ bb,
                              float* __restrict__ outf,
                              unsigned short* __restrict__ outb, int G) {
  __shared__ float rs[256], rq[256];
  int R = blockIdx.x, t = threadIdx.x;
  float a0 = 0.f, a1 = 0.f, invG = 1.f / (float)G;
  for (int q = 0; q < G; ++q) {
    size_t r = (size_t)(R * G + q) * DN;
    float p0 = proj[r + t], p1 = proj[r + t + 256];
    float y0 = (p0 >= th[t]       ? p0 : 0.f) + fused[r + t];
    float y1 = (p1 >= th[t + 256] ? p1 : 0.f) + fused[r + t + 256];
    rs[t] = y0 + y1; rq[t] = y0 * y0 + y1 * y1;
    __syncthreads();
    for (int st = 128; st > 0; st >>= 1) {
      if (t < st) { rs[t] += rs[t + st]; rq[t] += rq[t + st]; }
      __syncthreads();
    }
    float m = rs[0] * (1.f / DN);
    float var = rq[0] * (1.f / DN) - m * m;
    float inv = rsqrtf(var + 1e-5f);
    __syncthreads();
    a0 += ((y0 - m) * inv * g[t]       + bb[t])       * invG;
    a1 += ((y1 - m) * inv * g[t + 256] + bb[t + 256]) * invG;
  }
  size_t o = (size_t)R * DN;
  outf[o + t] = a0;        outf[o + t + 256] = a1;
  outb[o + t] = f2bf(a0);  outb[o + t + 256] = f2bf(a1);
}

// -------------------- binary-tree LIF over K then V ------------------------
// one block per (b,h); states carried K-pass -> V-pass in global scratch.
__global__ void k_tree(const float* __restrict__ qkv,
                       const float* __restrict__ nw,
                       const float* __restrict__ nbias,
                       const float* __restrict__ th,
                       const float* __restrict__ tau,
                       const float* __restrict__ vr,
                       float* __restrict__ states,
                       float* __restrict__ bufA, float* __restrict__ bufB,
                       float* __restrict__ tk, float* __restrict__ tv) {
  int bh = blockIdx.x;
  int b = bh >> 3, h = bh & 7;
  int t = threadIdx.x;
  const int lo_arr[4] = {7, 3, 1, 0};
  float* bufs[2] = { bufA + (size_t)bh * 16384, bufB + (size_t)bh * 16384 };

  for (int pass = 0; pass < 2; ++pass) {
    int col0 = ((pass + 1) * HN + h) * HDN;        // K: comp=1, V: comp=2
    float* root = (pass == 0 ? tk : tv) + (size_t)bh * 2048;
    for (int lvl = 0; lvl < 4; ++lvl) {
      int nodes = 8 >> lvl, lo = lo_arr[lvl];
      float* cur = bufs[lvl & 1];
      const float* prev = bufs[(lvl + 1) & 1];
      int total = nodes * 2048;
      for (int idx = t; idx < total; idx += 256) {
        int n = idx >> 11, rem = idx & 2047, s = rem >> 6, e = rem & 63;
        int ne = (lo + n) * HDN + e;
        float p = nbias[ne];
        const float* wrow = nw + (size_t)ne * HDN;
        if (lvl == 0) {
          const float* xin = qkv + (size_t)(b * LN_ + n * 32 + s) * NQ + col0;
#pragma unroll 8
          for (int d = 0; d < HDN; ++d) p += xin[d] * wrow[d];
        } else {
          const float* p0 = prev + (2 * n) * 2048 + s * 64;
          const float* p1 = prev + (2 * n + 1) * 2048 + s * 64;
#pragma unroll 8
          for (int d = 0; d < HDN; ++d) p += 0.5f * (p0[d] + p1[d]) * wrow[d];
        }
        size_t sti = ((size_t)bh * 15 + lo + n) * 2048 + s * 64 + e;
        float v  = (pass == 0) ? p : tau[ne] * states[sti] + p;
        float sp = (v >= th[ne]) ? 1.f : 0.f;
        if (pass == 0) states[sti] = v * (1.f - sp) + vr[ne] * sp;
        float o = p * sp;
        if (lvl < 3) cur[idx] = o; else root[rem] = o;
      }
      __syncthreads();
    }
  }
}

// --------------------------- attention over tree ---------------------------
// grid = B*H*(L/256); one q row per thread; K/V tiles in LDS.
__global__ void k_attn(const float* __restrict__ qkv,
                       const float* __restrict__ tk,
                       const float* __restrict__ tv,
                       float* __restrict__ attn) {
  __shared__ float ks[2048], vs[2048];
  int gb = blockIdx.x;
  int half = gb & 1, bh = gb >> 1;
  int b = bh >> 3, h = bh & 7;
  int t = threadIdx.x;
  for (int i = t; i < 2048; i += 256) {
    ks[i] = tk[(size_t)bh * 2048 + i];
    vs[i] = tv[(size_t)bh * 2048 + i];
  }
  __syncthreads();
  int row = half * 256 + t;
  const float* qp = qkv + (size_t)(b * LN_ + row) * NQ + h * HDN;
  float ql[64];
#pragma unroll
  for (int d = 0; d < 64; ++d) ql[d] = qp[d];
  float sc[32], mx = -3.4e38f;
  for (int j = 0; j < 32; ++j) {
    float s = 0.f;
#pragma unroll
    for (int d = 0; d < 64; ++d) s += ql[d] * ks[j * 64 + d];
    s *= 0.125f;                 // 1/sqrt(64)
    sc[j] = s; mx = fmaxf(mx, s);
  }
  float se = 0.f;
  for (int j = 0; j < 32; ++j) { float e = __expf(sc[j] - mx); sc[j] = e; se += e; }
  float inv = 1.f / se;
  float* op = attn + (size_t)(b * LN_ + row) * DN + h * HDN;
  for (int e = 0; e < 64; ++e) {
    float o = 0.f;
    for (int j = 0; j < 32; ++j) o += sc[j] * vs[j * 64 + e];
    op[e] = o * inv;
  }
}

// --------------------------- residual add + LN -----------------------------
__global__ void k_add_ln(const float* __restrict__ a, const float* __restrict__ r,
                         const float* __restrict__ g, const float* __restrict__ bb,
                         float* __restrict__ out) {
  __shared__ float rs[256], rq[256];
  int R = blockIdx.x, t = threadIdx.x;
  size_t o = (size_t)R * DN;
  float y0 = a[o + t] + r[o + t];
  float y1 = a[o + t + 256] + r[o + t + 256];
  rs[t] = y0 + y1; rq[t] = y0 * y0 + y1 * y1;
  __syncthreads();
  for (int st = 128; st > 0; st >>= 1) {
    if (t < st) { rs[t] += rs[t + st]; rq[t] += rq[t + st]; }
    __syncthreads();
  }
  float m = rs[0] * (1.f / DN);
  float var = rq[0] * (1.f / DN) - m * m;
  float inv = rsqrtf(var + 1e-5f);
  out[o + t]       = (y0 - m) * inv * g[t]       + bb[t];
  out[o + t + 256] = (y1 - m) * inv * g[t + 256] + bb[t + 256];
}

// -------------------- linear interpolate L=512 -> S=4096 -------------------
__global__ void k_interp(const float* __restrict__ base, float* __restrict__ out) {
  size_t i = (size_t)blockIdx.x * 256 + threadIdx.x;  // B*S*D elements
  int d = (int)(i & 511);
  size_t sb = i >> 9;
  int s = (int)(sb & 4095);
  int b = (int)(sb >> 12);
  float src = fmaxf((s + 0.5f) * 0.125f - 0.5f, 0.f);
  int i0 = (int)src;
  float w1 = src - (float)i0;
  int i1 = min(i0 + 1, LN_ - 1);
  size_t r0 = ((size_t)b * LN_ + i0) * DN + d;
  size_t r1 = ((size_t)b * LN_ + i1) * DN + d;
  out[i] = base[r0] * (1.f - w1) + base[r1] * w1;
}

// ---------------------------------------------------------------------------
extern "C" void kernel_launch(void* const* d_in, const int* in_sizes, int n_in,
                              void* d_out, int out_size, void* d_ws, size_t ws_size,
                              hipStream_t stream) {
  (void)in_sizes; (void)n_in; (void)out_size; (void)ws_size;
  const int*   text   = (const int*)d_in[0];
  const float* pe     = (const float*)d_in[1];
  const float* id_w   = (const float*)d_in[2];
  const float* id_b   = (const float*)d_in[3];
  const float* eg     = (const float*)d_in[4];
  const float* ebeta  = (const float*)d_in[5];
  const float* sub_w  = (const float*)d_in[6];
  const float* sub_b  = (const float*)d_in[7];
  const float* sub_th = (const float*)d_in[8];
  const float* sub_ng = (const float*)d_in[11];
  const float* sub_nb = (const float*)d_in[12];
  const float* word_w = (const float*)d_in[13];
  const float* word_b = (const float*)d_in[14];
  const float* word_th= (const float*)d_in[15];
  const float* word_ng= (const float*)d_in[18];
  const float* word_nb= (const float*)d_in[19];
  const float* qkv_w  = (const float*)d_in[20];
  const float* qkv_b  = (const float*)d_in[21];
  const float* node_w = (const float*)d_in[22];
  const float* node_b = (const float*)d_in[23];
  const float* node_th= (const float*)d_in[24];
  const float* node_tau=(const float*)d_in[25];
  const float* node_vr= (const float*)d_in[26];
  const float* out_ng = (const float*)d_in[27];
  const float* out_nb = (const float*)d_in[28];
  float* outp = (float*)d_out;

  // ---- workspace layout (persistent region + phase-aliased region X) ----
  char* w = (char*)d_ws;
  size_t o = 0;
  auto alloc = [&](size_t bytes) { char* p = w + o; o += (bytes + 255) & ~(size_t)255; return p; };
  float*          fusedW_f = (float*)alloc(33554432);           // [B,512,D] f32
  unsigned short* fusedW_b = (unsigned short*)alloc(16777216);
  float*          projW    = (float*)alloc(33554432);
  float*          word_f   = (float*)alloc(33554432);
  unsigned short* word_bf  = (unsigned short*)alloc(16777216);
  unsigned short* wsub_b   = (unsigned short*)alloc(524288);
  unsigned short* wword_b  = (unsigned short*)alloc(524288);
  unsigned short* wqkv_b   = (unsigned short*)alloc(1572864);
  char* X = w + o;
  // Phase A (through the sub-stage)
  float*          fusedS_f = (float*)(X);                       // 134 MB
  unsigned short* fusedS_b = (unsigned short*)(X + 134217728);  //  67 MB
  float*          projS    = (float*)(X + 134217728 + 67108864);// 134 MB
  // Phase B (fusedS/projS dead after LIF-pool): reuse X
  float* qkv    = (float*)(X);                                  // 100 MB
  float* states = (float*)(X + 100663296);                      //  31.5 MB
  float* bufA   = (float*)(X + 132120576);                      //  16.8 MB
  float* bufB   = (float*)(X + 148897792);                      //  16.8 MB
  float* tree_k = (float*)(X + 165675008);                      //   2 MB
  float* tree_v = (float*)(X + 167772160);                      //   2 MB
  float* attn   = (float*)(X + 169869312);                      //  33.5 MB
  float* base   = (float*)(X + 203423744);                      //  33.5 MB

  // 1) weights -> bf16
  k_cvt_bf16<<<512, 256, 0, stream>>>(sub_w,  wsub_b,  DN * DN);
  k_cvt_bf16<<<512, 256, 0, stream>>>(word_w, wword_b, DN * DN);
  k_cvt_bf16<<<1536, 256, 0, stream>>>(qkv_w, wqkv_b,  3 * DN * DN);

  // 2) embed + LN + pool2
  k_embed_pool2<<<BN * (SN / 2), 256, 0, stream>>>(text, pe, id_w, id_b, eg,
                                                   ebeta, fusedS_f, fusedS_b);
  // 3) sub projection (WMMA)
  k_gemm_bf16<<<dim3(DN / 128, (BN * SN / 2) / 64), 256, 0, stream>>>(
      fusedS_b, wsub_b, sub_b, projS, BN * SN / 2, DN, DN);
  // 4) LIF + LN + pool4 -> fused_word
  k_lif_ln_pool<<<BN * LN_, 256, 0, stream>>>(projS, fusedS_f, sub_th, sub_ng,
                                              sub_nb, fusedW_f, fusedW_b, 4);
  // 5) word projection (WMMA)
  k_gemm_bf16<<<dim3(DN / 128, (BN * LN_) / 64), 256, 0, stream>>>(
      fusedW_b, wword_b, word_b, projW, BN * LN_, DN, DN);
  // 6) LIF + LN -> word
  k_lif_ln_pool<<<BN * LN_, 256, 0, stream>>>(projW, fusedW_f, word_th, word_ng,
                                              word_nb, word_f, word_bf, 1);
  // 7) qkv projection (WMMA)
  k_gemm_bf16<<<dim3(NQ / 128, (BN * LN_) / 64), 256, 0, stream>>>(
      word_bf, wqkv_b, qkv_b, qkv, BN * LN_, NQ, DN);
  // 8) tree over K then V (state carried)
  k_tree<<<BN * HN, 256, 0, stream>>>(qkv, node_w, node_b, node_th, node_tau,
                                      node_vr, states, bufA, bufB, tree_k, tree_v);
  // 9) attention
  k_attn<<<BN * HN * (LN_ / 256), 256, 0, stream>>>(qkv, tree_k, tree_v, attn);
  // 10) residual + LN
  k_add_ln<<<BN * LN_, 256, 0, stream>>>(attn, word_f, out_ng, out_nb, base);
  // 11) interpolate L -> S
  k_interp<<<(BN * SN * DN) / 256, 256, 0, stream>>>(base, outp);
}
